// GCNNodeClassifierNetwork_33990371181433
// MI455X (gfx1250) — matrix operand order repaired and checked by reading, not compiled, not run on hardware
//
#include <hip/hip_runtime.h>
#include <hip/hip_bf16.h>
#include <math.h>

typedef __attribute__((ext_vector_type(16))) __bf16 v16bf;
typedef __attribute__((ext_vector_type(8)))  float  v8f;
typedef __attribute__((ext_vector_type(4)))  unsigned int v4u;
typedef __attribute__((ext_vector_type(8)))  int    v8i;
typedef __attribute__((ext_vector_type(4)))  int    v4i;

#define N_NODES 8192
#define IN_DIM  256
#define HID     512
#define NCLS    256

// ---- CDNA5 data-mover feature detection (degrade gracefully) --------------
#if __has_builtin(__builtin_amdgcn_global_load_async_to_lds_b128)
#define HAVE_ASYNC 1
#else
#define HAVE_ASYNC 0
#endif
#if __has_builtin(__builtin_amdgcn_tensor_load_to_lds)
#define HAVE_TDM 1
#else
#define HAVE_TDM 0
#endif

// Address-space-qualified 128-bit vector types (diagnostic showed the builtin
// takes 'int __vector(4) __device__*' i.e. v4i in AS1; LDS side is AS3).
typedef __attribute__((address_space(1))) v4i v4i_glb;
typedef __attribute__((address_space(3))) v4i v4i_lds;

// ISA 10.2: generic LDS pointer's low 32 bits are the LDS byte offset.
__device__ __forceinline__ unsigned lds_off_of(const void* p) {
    return (unsigned)(unsigned long long)(size_t)p;
}

#if HAVE_ASYNC
__device__ __forceinline__ void async_copy_b128(const void* g, void* l) {
    __builtin_amdgcn_global_load_async_to_lds_b128(
        (v4i_glb*)(unsigned long long)(size_t)g,
        (v4i_lds*)lds_off_of(l), 0, 0);
}
__device__ __forceinline__ void wait_async0() {
#if __has_builtin(__builtin_amdgcn_s_wait_asynccnt)
    __builtin_amdgcn_s_wait_asynccnt(0);
#else
    asm volatile("s_wait_asynccnt 0x0" ::: "memory");
#endif
}
#endif

#if HAVE_TDM
// 2-D TDM tile load: tile (rowLen x numRows) of 2-byte elems, row stride
// `strideElems`, from gptr into LDS at lds_off. D# packing per ISA ch.8.
__device__ __forceinline__ void tdm_load_2d_bf16(unsigned lds_off, const void* gptr,
                                                 unsigned rowLen, unsigned numRows,
                                                 unsigned strideElems,
                                                 unsigned tdim0, unsigned tdim1) {
    unsigned long long ga = (unsigned long long)(size_t)gptr;
    v4u g0;
    g0.x = 1u;                                             // count=1, user mode
    g0.y = lds_off;                                        // lds_addr
    g0.z = (unsigned)ga;                                   // global_addr[31:0]
    g0.w = (unsigned)((ga >> 32) & 0x01FFFFFFull) | (2u << 30);  // addr[56:32], type=2
    v8i g1;
    g1[0] = (int)(1u << 16);                               // data_size=1 (2 bytes)
    g1[1] = (int)((tdim0 & 0xFFFFu) << 16);                // tensor_dim0[15:0]
    g1[2] = (int)((tdim0 >> 16) | ((tdim1 & 0xFFFFu) << 16)); // dim0 hi | dim1 lo
    g1[3] = (int)(((tdim1 >> 16) & 0xFFFFu) | (rowLen << 16)); // dim1 hi | tile_dim0
    g1[4] = (int)numRows;                                  // tile_dim1 (tile_dim2=0)
    g1[5] = (int)strideElems;                              // tensor_dim0_stride lo
    g1[6] = 0;
    g1[7] = 0;
    v4i z4 = {0, 0, 0, 0};
#if __clang_major__ >= 23
    v8i z8 = {0, 0, 0, 0, 0, 0, 0, 0};
    __builtin_amdgcn_tensor_load_to_lds(g0, g1, z4, z4, z8, 0);
#else
    __builtin_amdgcn_tensor_load_to_lds(g0, g1, z4, z4, 0);
#endif
}
#endif

// ---------------------------------------------------------------------------
// deg[j] = sum_i A[i][j] + 1 ; dis = rsqrt(deg) ; dis2 = 1/deg
// ---------------------------------------------------------------------------
__global__ void deg_partial_kernel(const float* __restrict__ A,
                                   float* __restrict__ partial) {
    int j = blockIdx.x * 256 + threadIdx.x;
    int i0 = blockIdx.y * 256;
    float s = 0.f;
    for (int i = 0; i < 256; ++i)
        s += A[(size_t)(i0 + i) * N_NODES + j];
    partial[(size_t)blockIdx.y * N_NODES + j] = s;
}

__global__ void deg_final_kernel(const float* __restrict__ partial,
                                 float* __restrict__ dis,
                                 float* __restrict__ dis2) {
    int j = blockIdx.x * 256 + threadIdx.x;
    float s = 1.0f;                                // self loop
    for (int sl = 0; sl < 32; ++sl)
        s += partial[(size_t)sl * N_NODES + j];
    dis[j]  = rsqrtf(s);
    dis2[j] = 1.0f / s;
}

// ---------------------------------------------------------------------------
// Elementwise fp32 -> bf16 (A is 0/1 so this is exact). 8 elems/thread.
// ---------------------------------------------------------------------------
__global__ void conv_bf16_kernel(const float* __restrict__ src,
                                 __bf16* __restrict__ dst) {
    size_t i = ((size_t)blockIdx.x * blockDim.x + threadIdx.x) * 8;
    float4 f0 = *reinterpret_cast<const float4*>(src + i);
    float4 f1 = *reinterpret_cast<const float4*>(src + i + 4);
    union { __bf16 h[8]; uint4 u; } o;
    o.h[0] = (__bf16)f0.x; o.h[1] = (__bf16)f0.y;
    o.h[2] = (__bf16)f0.z; o.h[3] = (__bf16)f0.w;
    o.h[4] = (__bf16)f1.x; o.h[5] = (__bf16)f1.y;
    o.h[6] = (__bf16)f1.z; o.h[7] = (__bf16)f1.w;
    *reinterpret_cast<uint4*>(dst + i) = o.u;
}

// ---------------------------------------------------------------------------
// Transpose fp32 [R x C] -> bf16 [C x R], 32x32 LDS tiles.
// grid = (C/32, R/32), block = (32, 8)
// ---------------------------------------------------------------------------
__global__ void transpose_bf16_kernel(const float* __restrict__ src,
                                      __bf16* __restrict__ dst,
                                      int R, int C) {
    __shared__ float t[32][33];
    int tx = threadIdx.x, ty = threadIdx.y;
    int c0 = blockIdx.x * 32, r0 = blockIdx.y * 32;
#pragma unroll
    for (int k = 0; k < 4; ++k)
        t[ty + k * 8][tx] = src[(size_t)(r0 + ty + k * 8) * C + c0 + tx];
    __syncthreads();
#pragma unroll
    for (int k = 0; k < 4; ++k)
        dst[(size_t)(c0 + ty + k * 8) * R + r0 + tx] = (__bf16)t[tx][ty + k * 8];
}

// ---------------------------------------------------------------------------
// WMMA bf16 GEMM: C[M x N] = Abf[M x K] @ Bbf[K x N], row-major, f32 accum.
// Block: 256 threads (8 waves), 128x128 tile, K-step 32.
// A tile staged by TDM (tensor_load_to_lds) when available; B tile staged by
// async global->LDS copies into a row-major buffer, then transposed to [n][k].
// ---------------------------------------------------------------------------
__global__ __launch_bounds__(256) void wmma_gemm_bf16(
    const __bf16* __restrict__ Abf, const __bf16* __restrict__ Bbf,
    float* __restrict__ C, int M, int Nn, int K) {
    __shared__ __bf16 sA[128 * 32];    // [m][k]
    __shared__ __bf16 sB[128 * 32];    // transposed: [n][k]
#if HAVE_ASYNC
    __shared__ __bf16 sBst[32 * 128];  // row-major staging [k][n]
#endif

    const int tid  = threadIdx.x;
    const int lane = tid & 31;
    const int wave = tid >> 5;
    const int waveM = wave >> 2;       // 0..1
    const int waveN = wave & 3;        // 0..3
    const int mb = blockIdx.y * 128;
    const int nb = blockIdx.x * 128;
    const int half = lane >> 4;        // K-half select per ISA layout
    const int l16  = lane & 15;

    v8f acc[4][2] = {};

    const int ktiles = K >> 5;
    for (int kt = 0; kt < ktiles; ++kt) {
        const int k0 = kt << 5;

        // ---- A tile: 128 rows x 32 k (bf16, row-major in LDS) -------------
#if HAVE_TDM
        if (wave == 0)
            tdm_load_2d_bf16(lds_off_of(sA), &Abf[(size_t)mb * K + k0],
                             32u, 128u, (unsigned)K, (unsigned)K, (unsigned)M);
#elif HAVE_ASYNC
#pragma unroll
        for (int p = 0; p < 2; ++p) {
            int s = tid * 2 + p, row = s >> 2, kq = (s & 3) << 3;
            async_copy_b128(&Abf[(size_t)(mb + row) * K + k0 + kq],
                            &sA[row * 32 + kq]);
        }
#else
#pragma unroll
        for (int p = 0; p < 2; ++p) {
            int s = tid * 2 + p, row = s >> 2, kq = (s & 3) << 3;
            uint4 v = *reinterpret_cast<const uint4*>(
                &Abf[(size_t)(mb + row) * K + k0 + kq]);
            *reinterpret_cast<uint4*>(&sA[row * 32 + kq]) = v;
        }
#endif

        // ---- B tile: 32 k-rows x 128 n --------------------------------------
#if HAVE_ASYNC
#pragma unroll
        for (int p = 0; p < 2; ++p) {
            int s = tid * 2 + p, row = s >> 4, nq = (s & 15) << 3;
            async_copy_b128(&Bbf[(size_t)(k0 + row) * Nn + nb + nq],
                            &sBst[row * 128 + nq]);
        }
#else
#pragma unroll
        for (int p = 0; p < 2; ++p) {
            int s = tid * 2 + p, row = s >> 4, nq = (s & 15) << 3;
            union { uint4 u; __bf16 h[8]; } v;
            v.u = *reinterpret_cast<const uint4*>(
                &Bbf[(size_t)(k0 + row) * Nn + nb + nq]);
#pragma unroll
            for (int j = 0; j < 8; ++j)
                sB[(nq + j) * 32 + row] = v.h[j];
        }
#endif

        // ---- completion of the data movers ---------------------------------
#if HAVE_TDM
        if (wave == 0) __builtin_amdgcn_s_wait_tensorcnt((short)0);
#endif
#if HAVE_ASYNC
        wait_async0();
#endif
        __syncthreads();

#if HAVE_ASYNC
        // LDS->LDS transpose of the staged B tile into [n][k]
#pragma unroll
        for (int p = 0; p < 2; ++p) {
            int s = tid * 2 + p, row = s >> 4, nq = (s & 15) << 3;
            union { uint4 u; __bf16 h[8]; } v;
            v.u = *reinterpret_cast<const uint4*>(&sBst[row * 128 + nq]);
#pragma unroll
            for (int j = 0; j < 8; ++j)
                sB[(nq + j) * 32 + row] = v.h[j];
        }
        __syncthreads();
#endif

        // ---- fragments + WMMA ----------------------------------------------
        union Frag { uint4 u[2]; v16bf v; };
        Frag afrag[4], bfrag[2];
#pragma unroll
        for (int tm = 0; tm < 4; ++tm) {
            int row = waveM * 64 + tm * 16 + l16;
            int kc  = half * 8;    // lanes 0-15: K 0-7/16-23; 16-31: 8-15/24-31
            afrag[tm].u[0] = *reinterpret_cast<const uint4*>(&sA[row * 32 + kc]);
            afrag[tm].u[1] = *reinterpret_cast<const uint4*>(&sA[row * 32 + kc + 16]);
        }
#pragma unroll
        for (int tn = 0; tn < 2; ++tn) {
            int n  = waveN * 32 + tn * 16 + l16;
            int kc = half * 8;     // lanes 0-15: K 0-7/16-23; 16-31: 8-15/24-31
            bfrag[tn].u[0] = *reinterpret_cast<const uint4*>(&sB[n * 32 + kc]);
            bfrag[tn].u[1] = *reinterpret_cast<const uint4*>(&sB[n * 32 + kc + 16]);
        }
#pragma unroll
        for (int tm = 0; tm < 4; ++tm)
#pragma unroll
            for (int tn = 0; tn < 2; ++tn)
                acc[tm][tn] = __builtin_amdgcn_wmma_f32_16x16x32_bf16(
                    false, afrag[tm].v, false, bfrag[tn].v,
                    (short)0, acc[tm][tn], false, false);
        __syncthreads();
    }

    // store f32 C per ISA C/D layout: VGPR r -> M = (half*8 + r), lane&15 -> N
#pragma unroll
    for (int tm = 0; tm < 4; ++tm) {
        int m0 = mb + waveM * 64 + tm * 16 + half * 8;
#pragma unroll
        for (int tn = 0; tn < 2; ++tn) {
            int n = nb + waveN * 32 + tn * 16 + l16;
#pragma unroll
            for (int r = 0; r < 8; ++r)
                C[(size_t)(m0 + r) * Nn + n] = acc[tm][tn][r];
        }
    }
}

// ---------------------------------------------------------------------------
// m'[f][n] = bf16( dis[n] * h'[f][n] )   (feature-major, n fastest)
// ---------------------------------------------------------------------------
__global__ void scale_rows_kernel(const float* __restrict__ h,
                                  const float* __restrict__ dis,
                                  __bf16* __restrict__ out) {
    int idx = blockIdx.x * 256 + threadIdx.x;
    int n = idx & (N_NODES - 1);
    out[idx] = (__bf16)(h[idx] * dis[n]);
}

// ---------------------------------------------------------------------------
// hrelu'[f][n] = bf16( relu( dis[n]*agg + dis2[n]*hlin + b1[f] ) )
// ---------------------------------------------------------------------------
__global__ void epilogue_relu_kernel(const float* __restrict__ agg,
                                     const float* __restrict__ hlin,
                                     const float* __restrict__ dis,
                                     const float* __restrict__ dis2,
                                     const float* __restrict__ bias,
                                     __bf16* __restrict__ out) {
    int idx = blockIdx.x * 256 + threadIdx.x;
    int n = idx & (N_NODES - 1);
    int f = idx >> 13;
    float v = dis[n] * agg[idx] + dis2[n] * hlin[idx] + bias[f];
    out[idx] = (__bf16)fmaxf(v, 0.f);
}

// ---------------------------------------------------------------------------
// Final: out[n][f] = sigmoid( sp * (dis[n]*agg2'[f][n] + dis2[n]*h2'[f][n]
//                                   + b2[f] + x[n][f]) ), LDS transpose.
// ---------------------------------------------------------------------------
__global__ void final_kernel(const float* __restrict__ agg,
                             const float* __restrict__ h2,
                             const float* __restrict__ dis,
                             const float* __restrict__ dis2,
                             const float* __restrict__ b2,
                             const float* __restrict__ x,
                             const float* __restrict__ sp,
                             float* __restrict__ out) {
    __shared__ float tile[16][17];
    int tx = threadIdx.x, ty = threadIdx.y;
    int n0 = blockIdx.x * 16, f0 = blockIdx.y * 16;
    int n = n0 + tx, f = f0 + ty;
    size_t idx = (size_t)f * N_NODES + n;
    tile[ty][tx] = dis[n] * agg[idx] + dis2[n] * h2[idx] + b2[f];
    __syncthreads();
    int n2 = n0 + ty, f2 = f0 + tx;
    float s = sp[0] * (tile[tx][ty] + x[(size_t)n2 * NCLS + f2]);
    out[(size_t)n2 * NCLS + f2] = 1.0f / (1.0f + __expf(-s));
}

// ---------------------------------------------------------------------------
extern "C" void kernel_launch(void* const* d_in, const int* in_sizes, int n_in,
                              void* d_out, int out_size, void* d_ws, size_t ws_size,
                              hipStream_t stream) {
    const float* A  = (const float*)d_in[0];
    const float* x  = (const float*)d_in[1];
    const float* W1 = (const float*)d_in[2];
    const float* b1 = (const float*)d_in[3];
    const float* W2 = (const float*)d_in[4];
    const float* b2 = (const float*)d_in[5];
    const float* sp = (const float*)d_in[6];
    float* out = (float*)d_out;

    char* ws = (char*)d_ws;
    size_t off = 0;
    auto alloc = [&](size_t bytes) -> void* {
        void* p = ws + off;
        off = (off + bytes + 255) & ~(size_t)255;
        return p;
    };
    __bf16* Abf   = (__bf16*)alloc((size_t)N_NODES * N_NODES * 2);   // 128 MB
    float*  part  = (float*) alloc((size_t)32 * N_NODES * 4);
    float*  dis   = (float*) alloc((size_t)N_NODES * 4);
    float*  dis2  = (float*) alloc((size_t)N_NODES * 4);
    __bf16* xT    = (__bf16*)alloc((size_t)IN_DIM * N_NODES * 2);
    __bf16* W1T   = (__bf16*)alloc((size_t)HID * IN_DIM * 2);
    __bf16* W2T   = (__bf16*)alloc((size_t)NCLS * HID * 2);
    float*  h1    = (float*) alloc((size_t)HID * N_NODES * 4);
    float*  agg1  = (float*) alloc((size_t)HID * N_NODES * 4);
    __bf16* m1    = (__bf16*)alloc((size_t)HID * N_NODES * 2);
    __bf16* hre   = (__bf16*)alloc((size_t)HID * N_NODES * 2);
    float*  h2    = (float*) alloc((size_t)NCLS * N_NODES * 4);
    __bf16* m2    = (__bf16*)alloc((size_t)NCLS * N_NODES * 2);
    float*  agg2  = (float*) alloc((size_t)NCLS * N_NODES * 4);

    deg_partial_kernel<<<dim3(32, 32), 256, 0, stream>>>(A, part);
    deg_final_kernel<<<32, 256, 0, stream>>>(part, dis, dis2);

    conv_bf16_kernel<<<(N_NODES * (size_t)N_NODES) / (256 * 8), 256, 0, stream>>>(A, Abf);

    transpose_bf16_kernel<<<dim3(IN_DIM / 32, N_NODES / 32), dim3(32, 8), 0, stream>>>(x,  xT,  N_NODES, IN_DIM);
    transpose_bf16_kernel<<<dim3(HID / 32,    IN_DIM / 32),  dim3(32, 8), 0, stream>>>(W1, W1T, IN_DIM,  HID);
    transpose_bf16_kernel<<<dim3(NCLS / 32,   HID / 32),     dim3(32, 8), 0, stream>>>(W2, W2T, HID,     NCLS);

    // h1' = W1T @ xT   [512 x 8192], K=256
    wmma_gemm_bf16<<<dim3(N_NODES / 128, HID / 128), 256, 0, stream>>>(W1T, xT, h1, HID, N_NODES, IN_DIM);
    scale_rows_kernel<<<(HID * N_NODES) / 256, 256, 0, stream>>>(h1, dis, m1);
    // agg1' = m1' @ A   [512 x 8192], K=8192  (== (A.T @ m1).T)
    wmma_gemm_bf16<<<dim3(N_NODES / 128, HID / 128), 256, 0, stream>>>(m1, Abf, agg1, HID, N_NODES, N_NODES);
    epilogue_relu_kernel<<<(HID * N_NODES) / 256, 256, 0, stream>>>(agg1, h1, dis, dis2, b1, hre);

    // h2' = W2T @ hrelu'   [256 x 8192], K=512
    wmma_gemm_bf16<<<dim3(N_NODES / 128, NCLS / 128), 256, 0, stream>>>(W2T, hre, h2, NCLS, N_NODES, HID);
    scale_rows_kernel<<<(NCLS * N_NODES) / 256, 256, 0, stream>>>(h2, dis, m2);
    // agg2' = m2' @ A   [256 x 8192], K=8192
    wmma_gemm_bf16<<<dim3(N_NODES / 128, NCLS / 128), 256, 0, stream>>>(m2, Abf, agg2, NCLS, N_NODES, N_NODES);

    final_kernel<<<dim3(N_NODES / 16, NCLS / 16), dim3(16, 16), 0, stream>>>(
        agg2, h2, dis, dis2, b2, x, sp, out);
}
